// TemporalGraphTransformer_47141561041018
// MI455X (gfx1250) — matrix operand order repaired and checked by reading, not compiled, not run on hardware
//
#include <hip/hip_runtime.h>
#include <hip/hip_bf16.h>
#include <math.h>

#define NGRAPHS 64
#define NLAYERS 3
#define HEADS 4
#define HID 64
#define DMODEL 256
#define NETYPES 8
#define NCLASSES 2
#define NN 30000
#define NE 240000
#define QKVC 768   // concat q|k|v output columns

typedef __attribute__((ext_vector_type(16))) __bf16 v16bf;
typedef __attribute__((ext_vector_type(8)))  float  v8f;
typedef __attribute__((ext_vector_type(4)))  unsigned int v4u;

union BFrag { v16bf v; v4u u[2]; };

__device__ __forceinline__ unsigned short f2bf(float f) {
  unsigned int u = __float_as_uint(f);
  u += 0x7FFFu + ((u >> 16) & 1u);   // round-to-nearest-even
  return (unsigned short)(u >> 16);
}

__device__ __forceinline__ void atomicMaxF(float* addr, float val) {
  if (__float_as_int(val) >= 0)
    atomicMax((int*)addr, __float_as_int(val));
  else
    atomicMin((unsigned int*)addr, __float_as_uint(val));
}

// ---------------- weight pack: Wt[l][n][k] = W[l][k][n] (bf16), bias concat ----
__global__ void pack_weights_kernel(const float* __restrict__ Wq, const float* __restrict__ Wk,
                                    const float* __restrict__ Wv, const float* __restrict__ bq,
                                    const float* __restrict__ bk, const float* __restrict__ bv,
                                    unsigned short* __restrict__ wt, float* __restrict__ biascat) {
  long idx = blockIdx.x * (long)blockDim.x + threadIdx.x;
  const long total = (long)NLAYERS * QKVC * DMODEL;
  if (idx >= total) return;
  int k = (int)(idx % DMODEL);
  int n = (int)((idx / DMODEL) % QKVC);
  int l = (int)(idx / ((long)DMODEL * QKVC));
  int sel = n >> 8;         // 0:q 1:k 2:v
  int col = n & 255;
  const float* W = (sel == 0) ? Wq : (sel == 1) ? Wk : Wv;
  float w = W[((long)l * DMODEL + k) * DMODEL + col];
  wt[idx] = f2bf(w);
  if (k == 0) {
    const float* B = (sel == 0) ? bq : (sel == 1) ? bk : bv;
    biascat[l * QKVC + n] = B[l * DMODEL + col];
  }
}

// ---------------- x (f32) -> bf16 ----------------
__global__ void convert_x_kernel(const float* __restrict__ x, unsigned short* __restrict__ xbf) {
  long idx = blockIdx.x * (long)blockDim.x + threadIdx.x;
  if (idx >= (long)NN * DMODEL) return;
  xbf[idx] = f2bf(x[idx]);
}

// ---------------- per-layer scratch init ----------------
__global__ void init_layer_kernel(float* __restrict__ agg, float* __restrict__ mbuf,
                                  float* __restrict__ den) {
  long idx = blockIdx.x * (long)blockDim.x + threadIdx.x;
  if (idx < (long)NN * DMODEL) agg[idx] = 0.0f;
  if (idx < (long)NN * HEADS) { mbuf[idx] = -1e30f; den[idx] = 0.0f; }
}

// ---------------- fused QKV GEMM: [NN,256]bf16 x [256,768]bf16 -> [NN,768]f32 ----
// One wave computes a 16x64 output strip: 4 accumulators, 8 K-steps, 32 WMMAs.
__global__ void gemm_qkv_kernel(const unsigned short* __restrict__ xbf,
                                const unsigned short* __restrict__ wt,      // [768][256] bf16
                                const float* __restrict__ biascat,          // [768]
                                float* __restrict__ qkv) {
  const int lane = threadIdx.x & 31;
  const int wave = blockIdx.x * (blockDim.x >> 5) + (threadIdx.x >> 5);
  const int NROWT = NN / 16;        // 1875
  const int NCGRP = QKVC / 64;      // 12
  if (wave >= NROWT * NCGRP) return;
  const int rowTile  = wave / NCGRP;
  const int colGroup = wave % NCGRP;
  const int half = lane & 15;
  const int hi   = lane >> 4;

  v8f acc0 = {}, acc1 = {}, acc2 = {}, acc3 = {};
  const unsigned short* arow = xbf + (size_t)(rowTile * 16 + half) * DMODEL;
  const unsigned short* b0 = wt + (size_t)(colGroup * 64 +  0 + half) * DMODEL;
  const unsigned short* b1 = wt + (size_t)(colGroup * 64 + 16 + half) * DMODEL;
  const unsigned short* b2 = wt + (size_t)(colGroup * 64 + 32 + half) * DMODEL;
  const unsigned short* b3 = wt + (size_t)(colGroup * 64 + 48 + half) * DMODEL;

  #pragma unroll
  for (int k0 = 0; k0 < DMODEL; k0 += 32) {
    if (k0 + 32 < DMODEL) __builtin_prefetch(arow + k0 + 32, 0, 0);
    BFrag a;   // A: 16 contiguous K per lane (lo lanes K0..15, hi lanes K16..31)
    a.u[0] = *(const v4u*)(arow + k0 + hi * 16);
    a.u[1] = *(const v4u*)(arow + k0 + hi * 16 + 8);
    const int kb = k0 + hi * 16;   // B: 16 contiguous K per lane
    BFrag b;
    b.u[0] = *(const v4u*)(b0 + kb); b.u[1] = *(const v4u*)(b0 + kb + 8);
    acc0 = __builtin_amdgcn_wmma_f32_16x16x32_bf16(false, a.v, false, b.v, (short)0, acc0, false, false);
    b.u[0] = *(const v4u*)(b1 + kb); b.u[1] = *(const v4u*)(b1 + kb + 8);
    acc1 = __builtin_amdgcn_wmma_f32_16x16x32_bf16(false, a.v, false, b.v, (short)0, acc1, false, false);
    b.u[0] = *(const v4u*)(b2 + kb); b.u[1] = *(const v4u*)(b2 + kb + 8);
    acc2 = __builtin_amdgcn_wmma_f32_16x16x32_bf16(false, a.v, false, b.v, (short)0, acc2, false, false);
    b.u[0] = *(const v4u*)(b3 + kb); b.u[1] = *(const v4u*)(b3 + kb + 8);
    acc3 = __builtin_amdgcn_wmma_f32_16x16x32_bf16(false, a.v, false, b.v, (short)0, acc3, false, false);
  }

  const int colBase = colGroup * 64 + half;
  const float bias0 = biascat[colBase];
  const float bias1 = biascat[colBase + 16];
  const float bias2 = biascat[colBase + 32];
  const float bias3 = biascat[colBase + 48];
  const int rbase = rowTile * 16 + hi * 8;   // C/D: vgpr r -> row r (+8 for hi lanes)
  #pragma unroll
  for (int r = 0; r < 8; ++r) {
    float* orow = qkv + (size_t)(rbase + r) * QKVC;
    orow[colBase]      = acc0[r] + bias0;
    orow[colBase + 16] = acc1[r] + bias1;
    orow[colBase + 32] = acc2[r] + bias2;
    orow[colBase + 48] = acc3[r] + bias3;
  }
}

// ---------------- edge scores + segment atomic max (one wave per edge) --------
__global__ void edge_scores_kernel(const float* __restrict__ qkv, const int* __restrict__ src,
                                   const int* __restrict__ dst, const int* __restrict__ etype,
                                   const float* __restrict__ eemb_l,
                                   float* __restrict__ scores, float* __restrict__ mbuf) {
  const int lane = threadIdx.x & 31;
  const int e = blockIdx.x * (blockDim.x >> 5) + (threadIdx.x >> 5);
  if (e >= NE) return;
  const int s = src[e], d = dst[e], t = etype[e];
  const float* qp = qkv + (size_t)d * QKVC + lane * 8;
  const float* kp = qkv + (size_t)s * QKVC + 256 + lane * 8;
  const float* ep = eemb_l + t * DMODEL + lane * 8;
  float p = 0.0f;
  #pragma unroll
  for (int i = 0; i < 8; ++i) p += qp[i] * (kp[i] + ep[i]);
  p += __shfl_down(p, 4, 8);
  p += __shfl_down(p, 2, 8);
  p += __shfl_down(p, 1, 8);
  if ((lane & 7) == 0) {
    const int h = lane >> 3;
    float sc = p * 0.125f;   // 1/sqrt(HID)
    scores[(size_t)e * HEADS + h] = sc;
    atomicMaxF(mbuf + (size_t)d * HEADS + h, sc);
  }
}

// ---------------- exp(score - max), segment sum of exps ----------------------
__global__ void edge_expnorm_kernel(const int* __restrict__ dst, const float* __restrict__ mbuf,
                                    float* __restrict__ scores, float* __restrict__ den) {
  int idx = blockIdx.x * blockDim.x + threadIdx.x;
  if (idx >= NE * HEADS) return;
  int e = idx >> 2, h = idx & 3;
  int d = dst[e];
  float ex = __expf(scores[idx] - mbuf[(size_t)d * HEADS + h]);
  scores[idx] = ex;
  atomicAdd(den + (size_t)d * HEADS + h, ex);
}

// ---------------- weighted aggregation (one wave per edge) -------------------
__global__ void edge_aggregate_kernel(const float* __restrict__ qkv, const int* __restrict__ src,
                                      const int* __restrict__ dst, const int* __restrict__ etype,
                                      const float* __restrict__ eemb_l,
                                      const float* __restrict__ scores, const float* __restrict__ den,
                                      float* __restrict__ agg) {
  const int lane = threadIdx.x & 31;
  const int e = blockIdx.x * (blockDim.x >> 5) + (threadIdx.x >> 5);
  if (e >= NE) return;
  const int s = src[e], d = dst[e], t = etype[e];
  const int h = lane >> 3;
  float alpha = scores[(size_t)e * HEADS + h] / (den[(size_t)d * HEADS + h] + 1e-16f);
  const float* vp = qkv + (size_t)s * QKVC + 512 + lane * 8;
  const float* ep = eemb_l + t * DMODEL + lane * 8;
  float* ap = agg + (size_t)d * DMODEL + lane * 8;
  #pragma unroll
  for (int i = 0; i < 8; ++i) atomicAdd(ap + i, alpha * (vp[i] + ep[i]));
}

// ---------------- ELU, write f32 + bf16 copies ------------------------------
__global__ void node_elu_kernel(const float* __restrict__ agg, float* __restrict__ xf,
                                unsigned short* __restrict__ xbf) {
  long idx = blockIdx.x * (long)blockDim.x + threadIdx.x;
  if (idx >= (long)NN * DMODEL) return;
  float a = agg[idx];
  float y = (a > 0.0f) ? a : expm1f(a);
  xf[idx]  = y;
  xbf[idx] = f2bf(y);
}

// ---------------- graph mean pool ----------------
__global__ void pool_init_kernel(float* __restrict__ g, float* __restrict__ cnt) {
  int idx = blockIdx.x * blockDim.x + threadIdx.x;
  if (idx < NGRAPHS * DMODEL) g[idx] = 0.0f;
  if (idx < NGRAPHS) cnt[idx] = 0.0f;
}

__global__ void pool_kernel(const float* __restrict__ xf, const int* __restrict__ batch,
                            float* __restrict__ g, float* __restrict__ cnt) {
  long idx = blockIdx.x * (long)blockDim.x + threadIdx.x;
  if (idx >= (long)NN * DMODEL) return;
  int n = (int)(idx / DMODEL);
  int c = (int)(idx % DMODEL);
  int b = batch[n];
  atomicAdd(g + (size_t)b * DMODEL + c, xf[idx]);
  if (c == 0) atomicAdd(cnt + b, 1.0f);
}

// ---------------- GRU step (h0=0) + FC; one block per graph ------------------
__global__ void gru_fc_kernel(const float* __restrict__ g, const float* __restrict__ cnt,
                              const float* __restrict__ W_ih, const float* __restrict__ b_ih,
                              const float* __restrict__ b_hh, const float* __restrict__ W_fc,
                              const float* __restrict__ b_fc, float* __restrict__ out) {
  __shared__ float gs[DMODEL];
  __shared__ float hs[HID];
  const int gr = blockIdx.x;
  const int j  = threadIdx.x;   // 0..63
  float inv = 1.0f / fmaxf(cnt[gr], 1.0f);
  for (int k = j; k < DMODEL; k += HID) gs[k] = g[(size_t)gr * DMODEL + k] * inv;
  __syncthreads();
  float ir = 0.0f, iz = 0.0f, in_ = 0.0f;
  const float* wr = W_ih + (size_t)j * DMODEL;
  const float* wz = W_ih + (size_t)(HID + j) * DMODEL;
  const float* wn = W_ih + (size_t)(2 * HID + j) * DMODEL;
  for (int k = 0; k < DMODEL; ++k) {
    float gv = gs[k];
    ir += gv * wr[k]; iz += gv * wz[k]; in_ += gv * wn[k];
  }
  ir  += b_ih[j];
  iz  += b_ih[HID + j];
  in_ += b_ih[2 * HID + j];
  float r = 1.0f / (1.0f + __expf(-(ir + b_hh[j])));          // hr = b_hh (h0=0)
  float z = 1.0f / (1.0f + __expf(-(iz + b_hh[HID + j])));
  float nn = tanhf(in_ + r * b_hh[2 * HID + j]);
  hs[j] = (1.0f - z) * nn;                                    // + z*h0 = 0
  __syncthreads();
  if (j < NCLASSES) {
    float o = b_fc[j];
    for (int k = 0; k < HID; ++k) o += hs[k] * W_fc[j * HID + k];
    out[gr * NCLASSES + j] = o;
  }
}

extern "C" void kernel_launch(void* const* d_in, const int* in_sizes, int n_in,
                              void* d_out, int out_size, void* d_ws, size_t ws_size,
                              hipStream_t stream) {
  const float* x     = (const float*)d_in[0];
  const int*   ei    = (const int*)d_in[1];
  const int*   batch = (const int*)d_in[2];
  const int*   etype = (const int*)d_in[3];
  const float* Wq    = (const float*)d_in[4];
  const float* bq    = (const float*)d_in[5];
  const float* Wk    = (const float*)d_in[6];
  const float* bk    = (const float*)d_in[7];
  const float* Wv    = (const float*)d_in[8];
  const float* bv    = (const float*)d_in[9];
  const float* Eemb  = (const float*)d_in[10];
  const float* W_ih  = (const float*)d_in[11];
  const float* b_ih  = (const float*)d_in[12];
  // const float* W_hh = (const float*)d_in[13];   // unused: h0 == 0
  const float* b_hh  = (const float*)d_in[14];
  const float* W_fc  = (const float*)d_in[15];
  const float* b_fc  = (const float*)d_in[16];
  float* out = (float*)d_out;

  const int* src = ei;
  const int* dst = ei + NE;

  char* ws = (char*)d_ws;
  size_t off = 0;
  auto alloc = [&](size_t bytes) -> void* {
    void* p = ws + off;
    off = (off + bytes + 255) & ~(size_t)255;
    return p;
  };
  float*          qkv     = (float*)alloc((size_t)NN * QKVC * 4);          // 92.2 MB
  float*          agg     = (float*)alloc((size_t)NN * DMODEL * 4);        // 30.7 MB
  float*          xf      = (float*)alloc((size_t)NN * DMODEL * 4);        // 30.7 MB
  unsigned short* xbf     = (unsigned short*)alloc((size_t)NN * DMODEL * 2); // 15.4 MB
  unsigned short* wt      = (unsigned short*)alloc((size_t)NLAYERS * QKVC * DMODEL * 2);
  float*          biascat = (float*)alloc((size_t)NLAYERS * QKVC * 4);
  float*          scores  = (float*)alloc((size_t)NE * HEADS * 4);         // 3.8 MB
  float*          mbuf    = (float*)alloc((size_t)NN * HEADS * 4);
  float*          den     = (float*)alloc((size_t)NN * HEADS * 4);
  float*          gpool   = (float*)alloc((size_t)NGRAPHS * DMODEL * 4);
  float*          cnt     = (float*)alloc((size_t)NGRAPHS * 4);
  (void)ws_size; (void)in_sizes; (void)n_in; (void)out_size;

  // weight pack + x conversion
  {
    long total = (long)NLAYERS * QKVC * DMODEL;
    pack_weights_kernel<<<(unsigned)((total + 255) / 256), 256, 0, stream>>>(
        Wq, Wk, Wv, bq, bk, bv, wt, biascat);
    convert_x_kernel<<<(NN * DMODEL) / 256, 256, 0, stream>>>(x, xbf);
  }

  const int gemmWaves  = (NN / 16) * (QKVC / 64);       // 22500
  const int gemmBlocks = (gemmWaves + 7) / 8;           // 8 waves / block

  for (int l = 0; l < NLAYERS; ++l) {
    init_layer_kernel<<<(NN * DMODEL) / 256, 256, 0, stream>>>(agg, mbuf, den);
    gemm_qkv_kernel<<<gemmBlocks, 256, 0, stream>>>(
        xbf, wt + (size_t)l * QKVC * DMODEL, biascat + (size_t)l * QKVC, qkv);
    const float* eemb_l = Eemb + (size_t)l * NETYPES * DMODEL;
    edge_scores_kernel<<<NE / 8, 256, 0, stream>>>(qkv, src, dst, etype, eemb_l, scores, mbuf);
    edge_expnorm_kernel<<<(NE * HEADS) / 256, 256, 0, stream>>>(dst, mbuf, scores, den);
    edge_aggregate_kernel<<<NE / 8, 256, 0, stream>>>(qkv, src, dst, etype, eemb_l, scores, den, agg);
    node_elu_kernel<<<(NN * DMODEL) / 256, 256, 0, stream>>>(agg, xf, xbf);
  }

  pool_init_kernel<<<(NGRAPHS * DMODEL + 255) / 256, 256, 0, stream>>>(gpool, cnt);
  pool_kernel<<<(NN * DMODEL) / 256, 256, 0, stream>>>(xf, batch, gpool, cnt);
  gru_fc_kernel<<<NGRAPHS, HID, 0, stream>>>(gpool, cnt, W_ih, b_ih, b_hh, W_fc, b_fc, out);
}
